// RAR_44822278701751
// MI455X (gfx1250) — compile-verified, matrix operand order checked by reading, not run on hardware
//
#include <hip/hip_runtime.h>
#include <hip/hip_bf16.h>
#include <math.h>

// ---------------------------------------------------------------------------
// Problem constants (from reference)
// ---------------------------------------------------------------------------
#define T_STEPS 128
#define BATCH   256
#define EMBD    256
#define ACTD    32
#define HID     512
#define FEAT    512
#define XDIM    (EMBD + ACTD)   // 288, multiple of 32
#define WARM_UP 10
#define LOG_2PI 1.83787706640934548356f

typedef __attribute__((ext_vector_type(16))) __bf16 v16bf;
typedef __attribute__((ext_vector_type(8)))  float  v8f;

// CDNA5 async global->LDS copies (ASYNCcnt-tracked), guarded so the file
// still compiles if this toolchain doesn't declare the builtins.
#if defined(__has_builtin)
#if __has_builtin(__builtin_amdgcn_global_load_async_to_lds_b128) && \
    __has_builtin(__builtin_amdgcn_s_wait_asynccnt)
#define HAVE_ASYNC_LDS 1
#endif
#endif
#ifndef HAVE_ASYNC_LDS
#define HAVE_ASYNC_LDS 0
#endif

// Builtin signature (from hipcc diagnostic): param0 = AS1 pointer to
// 'int __attribute__((vector_size(16)))', param1 = AS3 pointer to same.
typedef int v4i_g __attribute__((vector_size(16)));
typedef __attribute__((address_space(1))) v4i_g* glob_v4i_ptr;
typedef __attribute__((address_space(3))) v4i_g* lds_v4i_ptr;

__device__ __forceinline__ void copy16_async(const __bf16* g, __bf16* l) {
#if HAVE_ASYNC_LDS
    __builtin_amdgcn_global_load_async_to_lds_b128(
        (glob_v4i_ptr)(void*)const_cast<__bf16*>(g),
        (lds_v4i_ptr)(void*)l,
        0, 0);
#else
    *(uint4*)l = *(const uint4*)g;
#endif
}

// ---------------------------------------------------------------------------
// bf16 WMMA GEMM:  Y[M,N] = act( X[M,K] @ W[N,K]^T + bias[N] )
//   ACT = 0 : linear   (store f32 and/or bf16)
//   ACT = 1 : ELU      (store f32 and/or bf16)
//   ACT = 2 : Gaussian loss epilogue: acc += sum 0.5*(Tgt - y)^2  (no store)
// Block tile 64x128, 128 threads = 4 waves in 2x2; each wave owns a 32x64
// quadrant = 2x4 v_wmma_f32_16x16x32_bf16 accumulators (8 WMMA / 6 fragment
// loads per K-chunk). Double-buffered LDS fed by async global->LDS copies.
// Requires: K % 32 == 0, M % 64 == 0, N % 128 == 0 (true for this workload).
// ---------------------------------------------------------------------------
template<int ACT>
__global__ __launch_bounds__(128)
void wmma_gemm_kernel(const __bf16* __restrict__ X, const __bf16* __restrict__ W,
                      const float* __restrict__ bias,
                      float* __restrict__ Yf, __bf16* __restrict__ Yb,
                      const float* __restrict__ Tgt, float* __restrict__ lossAcc,
                      int M, int N, int K)
{
    __shared__ __bf16 As[2][64][32];    // 2 x 4 KB
    __shared__ __bf16 Bs[2][128][32];   // 2 x 8 KB
    __shared__ float  red[128];

    const int tid  = threadIdx.x;
    const int lane = tid & 31;
    const int wave = tid >> 5;          // 0..3
    const int wr   = (wave >> 1) * 32;  // wave row offset inside 64x128 tile
    const int wc   = (wave & 1)  * 64;  // wave col offset

    const int m0 = blockIdx.y * 64;
    const int n0 = blockIdx.x * 128;

    v8f acc2[2][4];
    #pragma unroll
    for (int mt = 0; mt < 2; ++mt)
        #pragma unroll
        for (int nt = 0; nt < 4; ++nt)
            acc2[mt][nt] = (v8f){};

    const int laneM = lane & 15;
    const int kHalf = (lane < 16) ? 0 : 8;   // 16-bit A/B fragment K-half select

    // Stage one 64x32 A tile + 128x32 B tile into LDS buffer `buf`.
    // 6 copy instructions per thread (2 for A, 4 for B) => wait_asynccnt(6)
    // leaves exactly the next tile's copies in flight.
    auto stage = [&](int buf, int k0) {
        #pragma unroll
        for (int i = 0; i < 2; ++i) {
            int c = tid * 2 + i;              // 0..255
            int row = c >> 2, col = (c & 3) * 8;
            copy16_async(X + (size_t)(m0 + row) * K + k0 + col, &As[buf][row][col]);
        }
        #pragma unroll
        for (int i = 0; i < 4; ++i) {
            int c = tid * 4 + i;              // 0..511
            int row = c >> 2, col = (c & 3) * 8;
            copy16_async(W + (size_t)(n0 + row) * K + k0 + col, &Bs[buf][row][col]);
        }
    };

    const int nk = K / 32;
#if HAVE_ASYNC_LDS
    stage(0, 0);
#endif
    for (int it = 0; it < nk; ++it) {
        const int cur = it & 1;
#if HAVE_ASYNC_LDS
        const bool hasNext = (it + 1) < nk;
        if (hasNext) stage(cur ^ 1, (it + 1) * 32);      // overlap copy w/ compute
        if (hasNext) __builtin_amdgcn_s_wait_asynccnt(6); // current tile landed
        else         __builtin_amdgcn_s_wait_asynccnt(0);
        __syncthreads();                                  // all waves' copies done
#else
        stage(cur, it * 32);
        __syncthreads();
#endif
        // Fragments per documented 16-bit 16x32 layout:
        // lanes 0-15: K = {0..7,16..23}; lanes 16-31: K = {8..15,24..31}.
        v16bf a[2], b[4];
        #pragma unroll
        for (int i = 0; i < 16; ++i) {
            int kk = (i < 8) ? (kHalf + i) : (16 + kHalf + (i - 8));
            a[0][i] = As[cur][wr + laneM][kk];
            a[1][i] = As[cur][wr + 16 + laneM][kk];
            b[0][i] = Bs[cur][wc + laneM][kk];
            b[1][i] = Bs[cur][wc + 16 + laneM][kk];
            b[2][i] = Bs[cur][wc + 32 + laneM][kk];
            b[3][i] = Bs[cur][wc + 48 + laneM][kk];
        }
        #pragma unroll
        for (int mt = 0; mt < 2; ++mt)
            #pragma unroll
            for (int nt = 0; nt < 4; ++nt)
                acc2[mt][nt] = __builtin_amdgcn_wmma_f32_16x16x32_bf16(
                    false, a[mt], false, b[nt], (short)0, acc2[mt][nt], false, false);
        __syncthreads();   // safe to overwrite this buffer two iterations later
    }

    // Epilogue. C/D layout: VGPR j -> M = j (lanes 0-15) or j+8 (lanes 16-31),
    // N = lane & 15.
    float lsum = 0.f;
    const int mbase = m0 + wr + ((lane < 16) ? 0 : 8);

    auto epi = [&](int m, int n, float accv) {
        float y = accv + bias[n];
        if (ACT == 1) y = (y > 0.f) ? y : (expf(y) - 1.f);   // ELU
        if (ACT == 2) {
            float d = Tgt[(size_t)m * N + n] - y;
            lsum += 0.5f * d * d;
        } else {
            if (Yf) Yf[(size_t)m * N + n] = y;
            if (Yb) Yb[(size_t)m * N + n] = (__bf16)y;
        }
    };

    #pragma unroll
    for (int mt = 0; mt < 2; ++mt)
        #pragma unroll
        for (int nt = 0; nt < 4; ++nt) {
            const int ncol = n0 + wc + nt * 16 + laneM;
            #pragma unroll
            for (int j = 0; j < 8; ++j)
                epi(mbase + mt * 16 + j, ncol, acc2[mt][nt][j]);
        }

    if (ACT == 2) {
        red[tid] = lsum;
        __syncthreads();
        for (int s = 64; s > 0; s >>= 1) {
            if (tid < s) red[tid] += red[tid + s];
            __syncthreads();
        }
        if (tid == 0) atomicAdd(lossAcc, red[0]);
    }
}

// ---------------------------------------------------------------------------
// Elementwise helpers
// ---------------------------------------------------------------------------
__global__ void f32_to_bf16_kernel(const float* __restrict__ s, __bf16* __restrict__ d, int n) {
    int i = blockIdx.x * blockDim.x + threadIdx.x;
    if (i < n) d[i] = (__bf16)s[i];
}
__global__ void zero_f32_kernel(float* p, int n) {
    int i = blockIdx.x * blockDim.x + threadIdx.x;
    if (i < n) p[i] = 0.f;
}
__global__ void zero_bf16_kernel(__bf16* p, int n) {
    int i = blockIdx.x * blockDim.x + threadIdx.x;
    if (i < n) p[i] = (__bf16)0.f;
}

// x[b, 0:256] = (useMu ? mu : emb_t), x[b, 256:288] = zeroAction ? 0 : action_t
__global__ void concat_select_kernel(const float* __restrict__ emb_t,
                                     const float* __restrict__ action_t,
                                     const float* __restrict__ mu,
                                     int useMu, int zeroAction,
                                     __bf16* __restrict__ x)
{
    int idx = blockIdx.x * blockDim.x + threadIdx.x;
    if (idx >= BATCH * XDIM) return;
    int b = idx / XDIM, c = idx % XDIM;
    float v;
    if (c < EMBD) v = useMu ? mu[b * EMBD + c] : emb_t[b * EMBD + c];
    else          v = zeroAction ? 0.f : action_t[b * ACTD + (c - EMBD)];
    x[idx] = (__bf16)v;
}

// GRU gate fusion: h' = (1-z)*n + z*h  from gi[b,3H], gh[b,3H] (bias included)
__global__ void gru_fuse_kernel(const float* __restrict__ gi, const float* __restrict__ gh,
                                const float* __restrict__ h,
                                float* __restrict__ hnew, __bf16* __restrict__ hnew_bf)
{
    int idx = blockIdx.x * blockDim.x + threadIdx.x;
    if (idx >= BATCH * HID) return;
    int b = idx / HID, j = idx % HID;
    const float* gib = gi + (size_t)b * 3 * HID;
    const float* ghb = gh + (size_t)b * 3 * HID;
    float r = 1.f / (1.f + expf(-(gib[j]           + ghb[j])));
    float z = 1.f / (1.f + expf(-(gib[j + HID]     + ghb[j + HID])));
    float n = tanhf(gib[j + 2 * HID] + r * ghb[j + 2 * HID]);
    float out = (1.f - z) * n + z * h[idx];
    hnew[idx]    = out;
    hnew_bf[idx] = (__bf16)out;
}

// Reward head (N=1): one wave per row, wave-reduce dot, fused Gaussian loss.
__global__ __launch_bounds__(256)
void rew_dot_loss_kernel(const __bf16* __restrict__ X, const __bf16* __restrict__ w,
                         const float* __restrict__ bias, const float* __restrict__ tgt,
                         float* __restrict__ acc, int M)
{
    int row  = blockIdx.x * (blockDim.x >> 5) + (threadIdx.x >> 5);
    int lane = threadIdx.x & 31;
    if (row >= M) return;
    float s = 0.f;
    for (int k = lane; k < FEAT; k += 32)
        s += (float)X[(size_t)row * FEAT + k] * (float)w[k];
    for (int off = 16; off > 0; off >>= 1) s += __shfl_xor(s, off, 32);
    if (lane == 0) {
        float y = s + bias[0];
        float d = tgt[row] - y;
        atomicAdd(acc, 0.5f * d * d);
    }
}

__global__ void finalize_kernel(const float* __restrict__ acc, float* __restrict__ out) {
    if (blockIdx.x == 0 && threadIdx.x == 0)
        out[0] = acc[0] / (float)(T_STEPS * BATCH) + 0.5f * LOG_2PI * (float)(EMBD + 1);
}

// ---------------------------------------------------------------------------
// Host orchestration
// ---------------------------------------------------------------------------
static inline void launch_gemm(hipStream_t s, int act,
                               const __bf16* X, const __bf16* W, const float* bias,
                               float* Yf, __bf16* Yb, const float* tgt, float* acc,
                               int M, int N, int K)
{
    dim3 g(N / 128, M / 64), b(128);
    if (act == 0)      wmma_gemm_kernel<0><<<g, b, 0, s>>>(X, W, bias, Yf, Yb, tgt, acc, M, N, K);
    else if (act == 1) wmma_gemm_kernel<1><<<g, b, 0, s>>>(X, W, bias, Yf, Yb, tgt, acc, M, N, K);
    else               wmma_gemm_kernel<2><<<g, b, 0, s>>>(X, W, bias, Yf, Yb, tgt, acc, M, N, K);
}

static inline void cvt(hipStream_t s, const float* src, __bf16* dst, int n) {
    f32_to_bf16_kernel<<<(n + 255) / 256, 256, 0, s>>>(src, dst, n);
}

extern "C" void kernel_launch(void* const* d_in, const int* in_sizes, int n_in,
                              void* d_out, int out_size, void* d_ws, size_t ws_size,
                              hipStream_t stream) {
    (void)in_sizes; (void)n_in; (void)out_size; (void)ws_size;

    const float* emb    = (const float*)d_in[0];   // [T,B,256]
    const float* action = (const float*)d_in[1];   // [T,B,32]
    const float* reward = (const float*)d_in[2];   // [T,B]
    const float* gWih   = (const float*)d_in[3];   // [1536,288]
    const float* gWhh   = (const float*)d_in[4];   // [1536,512]
    const float* gbih   = (const float*)d_in[5];
    const float* gbhh   = (const float*)d_in[6];
    const float* eW1    = (const float*)d_in[7];   const float* eb1 = (const float*)d_in[8];
    const float* eW2    = (const float*)d_in[9];   const float* eb2 = (const float*)d_in[10];
    const float* eWo    = (const float*)d_in[11];  const float* ebo = (const float*)d_in[12];
    const float* rW1    = (const float*)d_in[13];  const float* rb1 = (const float*)d_in[14];
    const float* rW2    = (const float*)d_in[15];  const float* rb2 = (const float*)d_in[16];
    const float* rWo    = (const float*)d_in[17];  const float* rbo = (const float*)d_in[18];

    // ---- workspace bump allocator ----
    char* ws = (char*)d_ws;
    size_t off = 0;
    auto alloc = [&](size_t bytes) -> void* {
        void* p = ws + off;
        off += (bytes + 255) & ~(size_t)255;
        return p;
    };

    __bf16* wbWih = (__bf16*)alloc((size_t)3 * HID * XDIM * 2);
    __bf16* wbWhh = (__bf16*)alloc((size_t)3 * HID * HID * 2);
    __bf16* wbeW1 = (__bf16*)alloc((size_t)FEAT * HID * 2);
    __bf16* wbeW2 = (__bf16*)alloc((size_t)FEAT * FEAT * 2);
    __bf16* wbeWo = (__bf16*)alloc((size_t)EMBD * FEAT * 2);
    __bf16* wbrW1 = (__bf16*)alloc((size_t)FEAT * HID * 2);
    __bf16* wbrW2 = (__bf16*)alloc((size_t)FEAT * FEAT * 2);
    __bf16* wbrWo = (__bf16*)alloc((size_t)FEAT * 2);

    const size_t BH = (size_t)BATCH * HID;
    __bf16* states = (__bf16*)alloc((size_t)(T_STEPS + 1) * BH * 2); // bf16 h_t
    __bf16* st1    = (__bf16*)alloc(BH * 2);                         // step MLP tmp
    __bf16* st2    = (__bf16*)alloc(BH * 2);
    __bf16* xbuf   = (__bf16*)alloc((size_t)BATCH * XDIM * 2);       // GRU input
    __bf16* zerob  = (__bf16*)alloc(BH * 2);                         // h=0 (bf16)
    float*  hcur   = (float*)alloc(BH * 4);                          // h (f32)
    float*  gif    = (float*)alloc((size_t)BATCH * 3 * HID * 4);
    float*  ghf    = (float*)alloc((size_t)BATCH * 3 * HID * 4);
    float*  muf    = (float*)alloc((size_t)BATCH * EMBD * 4);
    float*  acc    = (float*)alloc(4);

    const int CH = 8192;                          // final-phase row chunk
    __bf16* t1 = (__bf16*)alloc((size_t)CH * FEAT * 2);
    __bf16* t2 = (__bf16*)alloc((size_t)CH * FEAT * 2);

    // ---- weight conversion + init ----
    cvt(stream, gWih, wbWih, 3 * HID * XDIM);
    cvt(stream, gWhh, wbWhh, 3 * HID * HID);
    cvt(stream, eW1,  wbeW1, FEAT * HID);
    cvt(stream, eW2,  wbeW2, FEAT * FEAT);
    cvt(stream, eWo,  wbeWo, EMBD * FEAT);
    cvt(stream, rW1,  wbrW1, FEAT * HID);
    cvt(stream, rW2,  wbrW2, FEAT * FEAT);
    cvt(stream, rWo,  wbrWo, FEAT);
    zero_f32_kernel<<<(int)(BH + 255) / 256, 256, 0, stream>>>(hcur, (int)BH);
    zero_bf16_kernel<<<(int)(BH + 255) / 256, 256, 0, stream>>>(zerob, (int)BH);
    zero_f32_kernel<<<1, 32, 0, stream>>>(acc, 1);

    // ---- initial state: h0 = GRU(h=0, [emb[0], 0]) ----
    concat_select_kernel<<<(BATCH * XDIM + 255) / 256, 256, 0, stream>>>(
        emb, action, muf, /*useMu=*/0, /*zeroAction=*/1, xbuf);
    launch_gemm(stream, 0, xbuf,  wbWih, gbih, gif, nullptr, nullptr, nullptr, BATCH, 3 * HID, XDIM);
    launch_gemm(stream, 0, zerob, wbWhh, gbhh, ghf, nullptr, nullptr, nullptr, BATCH, 3 * HID, HID);
    gru_fuse_kernel<<<(int)(BH + 255) / 256, 256, 0, stream>>>(gif, ghf, hcur, hcur, states);

    // ---- sequential scan over T ----
    for (int t = 0; t < T_STEPS; ++t) {
        const __bf16* hbf = states + (size_t)t * BH;   // current state (bf16)
        // teacher-forcing MLP: mu = MLP(state)
        launch_gemm(stream, 1, hbf, wbeW1, eb1, nullptr, st1, nullptr, nullptr, BATCH, FEAT, HID);
        launch_gemm(stream, 1, st1, wbeW2, eb2, nullptr, st2, nullptr, nullptr, BATCH, FEAT, FEAT);
        launch_gemm(stream, 0, st2, wbeWo, ebo, muf, nullptr, nullptr, nullptr, BATCH, EMBD, FEAT);
        // x = [t < WARM_UP ? emb[t] : mu, action[t]]
        concat_select_kernel<<<(BATCH * XDIM + 255) / 256, 256, 0, stream>>>(
            emb + (size_t)t * BATCH * EMBD, action + (size_t)t * BATCH * ACTD,
            muf, t >= WARM_UP ? 1 : 0, 0, xbuf);
        // GRU gates
        launch_gemm(stream, 0, xbuf, wbWih, gbih, gif, nullptr, nullptr, nullptr, BATCH, 3 * HID, XDIM);
        launch_gemm(stream, 0, hbf,  wbWhh, gbhh, ghf, nullptr, nullptr, nullptr, BATCH, 3 * HID, HID);
        gru_fuse_kernel<<<(int)(BH + 255) / 256, 256, 0, stream>>>(
            gif, ghf, hcur, hcur, states + (size_t)(t + 1) * BH);
    }

    // ---- batched heads over all T*B states (chunked), loss fused in epilogue ----
    const int MTOT = T_STEPS * BATCH;
    for (int c0 = 0; c0 < MTOT; c0 += CH) {
        const __bf16* Xs = states + (size_t)c0 * FEAT;
        // emb head -> Gaussian loss vs emb
        launch_gemm(stream, 1, Xs, wbeW1, eb1, nullptr, t1, nullptr, nullptr, CH, FEAT, HID);
        launch_gemm(stream, 1, t1, wbeW2, eb2, nullptr, t2, nullptr, nullptr, CH, FEAT, FEAT);
        launch_gemm(stream, 2, t2, wbeWo, ebo, nullptr, nullptr,
                    emb + (size_t)c0 * EMBD, acc, CH, EMBD, FEAT);
        // reward head -> Gaussian loss vs reward
        launch_gemm(stream, 1, Xs, wbrW1, rb1, nullptr, t1, nullptr, nullptr, CH, FEAT, HID);
        launch_gemm(stream, 1, t1, wbrW2, rb2, nullptr, t2, nullptr, nullptr, CH, FEAT, FEAT);
        rew_dot_loss_kernel<<<CH / 8, 256, 0, stream>>>(t2, wbrWo, rbo, reward + c0, acc, CH);
    }

    finalize_kernel<<<1, 32, 0, stream>>>(acc, (float*)d_out);
}
